// SelectiveSSM_10393820856794
// MI455X (gfx1250) — matrix operand order repaired and checked by reading, not compiled, not run on hardware
//
#include <hip/hip_runtime.h>
#include <math.h>

// ---------------------------------------------------------------------------
// Problem constants (from reference)
// ---------------------------------------------------------------------------
#define BATCH   2
#define SEQLEN  2048
#define DMODEL  512
#define DSTATE  16
#define DCONV   4
#define DINNER  1024          // DMODEL * EXPAND
#define NROWS   (BATCH * SEQLEN)        // 4096
#define N1      (2 * DINNER)            // 2048 (xz width)
#define PROJW   48                      // padded 42 -> 48
#define NCHUNK  64                      // chunks along L for parallel scan
#define CSIZE   (SEQLEN / NCHUNK)       // 32 steps per chunk
#define DGROUPS (DINNER / 128)          // 8

typedef __attribute__((ext_vector_type(16))) __bf16 v16bf;
typedef __attribute__((ext_vector_type(8)))  __bf16 v8bf;
typedef __attribute__((ext_vector_type(8)))  float  v8f;

static __device__ __forceinline__ float softplus_f(float x) {
    return (x > 20.0f) ? x : log1pf(expf(x));
}
static __device__ __forceinline__ float sigmoid_f(float x) {
    return 1.0f / (1.0f + expf(-x));
}

// ---------------------------------------------------------------------------
// f32 -> bf16 elementwise convert
// ---------------------------------------------------------------------------
__global__ void cvt_bf16_kernel(const float* __restrict__ in,
                                __bf16* __restrict__ out, int n) {
    int i = blockIdx.x * blockDim.x + threadIdx.x;
    if (i < n) out[i] = (__bf16)in[i];
}

// ---------------------------------------------------------------------------
// f32 (K x N) -> bf16 (N x K) transpose (weights, one-time per call)
// ---------------------------------------------------------------------------
__global__ void transpose_bf16_kernel(const float* __restrict__ in,
                                      __bf16* __restrict__ out, int K, int N) {
    int i = blockIdx.x * blockDim.x + threadIdx.x;
    if (i >= K * N) return;
    int k = i / N, n = i % N;
    out[(size_t)n * K + k] = (__bf16)in[(size_t)k * N + n];
}

// ---------------------------------------------------------------------------
// WMMA bf16 GEMM: C(MxN) = A(MxK) * B(KxN), B supplied transposed (NxK).
// One wave computes a 16x64 strip (4 accumulators): A fragment reused across
// 4 v_wmma_f32_16x16x32_bf16 per K-step -> 4x less A traffic, 4 independent
// WMMAs in flight to hide load latency.
// ---------------------------------------------------------------------------
__global__ __launch_bounds__(256) void gemm_bf16_wmma_kernel(
    const __bf16* __restrict__ A,    // M x K row-major
    const __bf16* __restrict__ BT,   // N x K row-major
    float* __restrict__ C,           // M x N row-major
    int M, int N, int K) {
    int gtid   = blockIdx.x * blockDim.x + threadIdx.x;
    int wave   = gtid >> 5;
    int lane   = threadIdx.x & 31;
    int strips = N >> 6;                 // groups of 4 N-tiles
    int tm = (wave / strips) << 4;
    int tn = (wave % strips) << 6;
    if (tm >= M) return;                 // wave-uniform: EXEC stays all-ones

    int half = lane >> 4;                // 0: lanes 0-15, 1: lanes 16-31
    int l16  = lane & 15;
    const __bf16* arow = A  + (size_t)(tm + l16) * K;
    const __bf16* b0   = BT + (size_t)(tn      + l16) * K;
    const __bf16* b1   = BT + (size_t)(tn + 16 + l16) * K;
    const __bf16* b2   = BT + (size_t)(tn + 32 + l16) * K;
    const __bf16* b3   = BT + (size_t)(tn + 48 + l16) * K;

    v8f acc0 = {}, acc1 = {}, acc2 = {}, acc3 = {};
    for (int k0 = 0; k0 < K; k0 += 32) {
        // A fragment: elems e -> K = k0 + 16*half + e (contiguous)
        v16bf a = *(const v16bf*)(arow + k0 + 16 * half);
        if (k0 + 64 < K)                 // uniform branch; streaming prefetch
            __builtin_prefetch((const void*)(arow + k0 + 64), 0, 0);
        int kb = k0 + 16 * half;         // B elems e -> K = kb + e (contiguous)
        acc0 = __builtin_amdgcn_wmma_f32_16x16x32_bf16(
            false, a, false, *(const v16bf*)(b0 + kb), (short)0, acc0, false, false);
        acc1 = __builtin_amdgcn_wmma_f32_16x16x32_bf16(
            false, a, false, *(const v16bf*)(b1 + kb), (short)0, acc1, false, false);
        acc2 = __builtin_amdgcn_wmma_f32_16x16x32_bf16(
            false, a, false, *(const v16bf*)(b2 + kb), (short)0, acc2, false, false);
        acc3 = __builtin_amdgcn_wmma_f32_16x16x32_bf16(
            false, a, false, *(const v16bf*)(b3 + kb), (short)0, acc3, false, false);
    }
    // D layout: VGPR r -> M = tm + r + 8*half, N = tn + l16 (+16*tile)
#pragma unroll
    for (int r = 0; r < 8; ++r) {
        size_t row = (size_t)(tm + r + 8 * half) * N + tn + l16;
        C[row]      = acc0[r];
        C[row + 16] = acc1[r];
        C[row + 32] = acc2[r];
        C[row + 48] = acc3[r];
    }
}

// ---------------------------------------------------------------------------
// Causal depthwise conv1d (+bias) + SiLU.  xz row stride = 2*DINNER,
// x_ssm part = columns [0, DINNER).
// ---------------------------------------------------------------------------
__global__ void conv_silu_kernel(const float* __restrict__ xz,
                                 const float* __restrict__ conv_w,
                                 const float* __restrict__ conv_b,
                                 float* __restrict__ x_ssm) {
    int i = blockIdx.x * blockDim.x + threadIdx.x;   // over NROWS*DINNER
    if (i >= NROWS * DINNER) return;
    int d = i % DINNER;
    int r = i / DINNER;                               // b*L + l
    int l = r % SEQLEN;
    float acc = conv_b[d];
#pragma unroll
    for (int t = 0; t < DCONV; ++t) {
        int ls = l - (DCONV - 1) + t;
        if (ls >= 0)
            acc += conv_w[d * DCONV + t] * xz[(size_t)(r - (l - ls)) * N1 + d];
    }
    x_ssm[i] = acc * sigmoid_f(acc);
}

// ---------------------------------------------------------------------------
// Fused small projection: per row r, 42 outputs over K=DINNER:
//   j in [0,33): x_ssm @ W_x   (DINNER x 33)
//   j in [33,41): x_ssm @ W_theta (DINNER x 8)
//   j == 41   : x_ssm @ W_lam (DINNER x 1)
// ---------------------------------------------------------------------------
__global__ __launch_bounds__(64) void proj_kernel(
    const float* __restrict__ x_ssm,
    const float* __restrict__ W_x,
    const float* __restrict__ W_theta,
    const float* __restrict__ W_lam,
    float* __restrict__ proj) {
    int r = blockIdx.x;
    int j = threadIdx.x;
    if (j >= 42) return;
    const float* xr = x_ssm + (size_t)r * DINNER;
    float acc = 0.0f;
    if (j < 33) {
        for (int k = 0; k < DINNER; ++k) acc += xr[k] * W_x[k * 33 + j];
    } else if (j < 41) {
        int jj = j - 33;
        for (int k = 0; k < DINNER; ++k) acc += xr[k] * W_theta[k * 8 + jj];
    } else {
        for (int k = 0; k < DINNER; ++k) acc += xr[k] * W_lam[k];
    }
    proj[(size_t)r * PROJW + j] = acc;
}

// ---------------------------------------------------------------------------
// cumsum of theta (8 lanes) along L, per batch
// ---------------------------------------------------------------------------
__global__ void cumsum_theta_kernel(const float* __restrict__ proj,
                                    float* __restrict__ theta_cum) {
    int b = blockIdx.x;          // BATCH
    int j = threadIdx.x;         // 8
    float run = 0.0f;
    for (int l = 0; l < SEQLEN; ++l) {
        size_t r = (size_t)b * SEQLEN + l;
        run += proj[r * PROJW + 33 + j];
        theta_cum[r * 8 + j] = run;
    }
}

// ---------------------------------------------------------------------------
// RoPE on B,C (+biases) and per-row scalars dt=softplus(.), lam=sigmoid(.)
// ---------------------------------------------------------------------------
__global__ void rope_kernel(const float* __restrict__ proj,
                            const float* __restrict__ theta_cum,
                            const float* __restrict__ B_bias,
                            const float* __restrict__ C_bias,
                            const float* __restrict__ b_lam,
                            float* __restrict__ Bm_rope,
                            float* __restrict__ Cm_rope,
                            float* __restrict__ scal) {
    int r = blockIdx.x * blockDim.x + threadIdx.x;
    if (r >= NROWS) return;
    float ct[8], st[8], Bm[16], Cm[16];
#pragma unroll
    for (int h = 0; h < 8; ++h) {
        float th = theta_cum[(size_t)r * 8 + h];
        ct[h] = cosf(th); st[h] = sinf(th);
    }
#pragma unroll
    for (int n = 0; n < 16; ++n) {
        Bm[n] = proj[(size_t)r * PROJW + n]      + B_bias[n];
        Cm[n] = proj[(size_t)r * PROJW + 16 + n] + C_bias[n];
    }
#pragma unroll
    for (int n = 0; n < 16; ++n) {
        int h = n & 7;
        float rb = (n < 8) ? -Bm[n + 8] : Bm[n - 8];
        float rc = (n < 8) ? -Cm[n + 8] : Cm[n - 8];
        Bm_rope[(size_t)r * 16 + n] = Bm[n] * ct[h] + rb * st[h];
        Cm_rope[(size_t)r * 16 + n] = Cm[n] * ct[h] + rc * st[h];
    }
    scal[(size_t)r * 2 + 0] = softplus_f(proj[(size_t)r * PROJW + 32]);
    scal[(size_t)r * 2 + 1] = sigmoid_f(proj[(size_t)r * PROJW + 41] + b_lam[0]);
}

// ---------------------------------------------------------------------------
// Chunked parallel scan.  h_t = alpha_t*h_{t-1} + inp_t is linear in the
// incoming state, so split L into NCHUNK chunks:
//   pass1: per chunk compute P = prod(alpha), q = chunk scan from h=0
//   pass2: tiny sequential combine over chunks -> h_start per chunk
//   pass3: re-scan each chunk from h_start, fuse skip + SiLU(z) gate -> bf16
// pass1/3 grids: BATCH*NCHUNK*DGROUPS = 1024 blocks of 128 (vs 16 before).
// ---------------------------------------------------------------------------
__global__ __launch_bounds__(128) void scan_pass1_kernel(
    const float* __restrict__ x_ssm,
    const float* __restrict__ Bm_rope,
    const float* __restrict__ scal,
    const float* __restrict__ A_log,
    const float* __restrict__ W_dt,
    const float* __restrict__ b_dt,
    float* __restrict__ Pbuf,
    float* __restrict__ Qbuf) {
    int g = blockIdx.x % DGROUPS;
    int c = (blockIdx.x / DGROUPS) % NCHUNK;
    int b = blockIdx.x / (DGROUPS * NCHUNK);
    int d = g * 128 + threadIdx.x;

    __shared__ float sB[16], sS[2], sBp[16];
    size_t r0 = (size_t)b * SEQLEN + c * CSIZE;
    if (threadIdx.x < 16)
        sBp[threadIdx.x] = (c == 0) ? 0.0f : Bm_rope[(r0 - 1) * 16 + threadIdx.x];

    float Arow[DSTATE], h[DSTATE], P[DSTATE], Bxp[DSTATE];
    float wdt = W_dt[d], bdt = b_dt[d];
#pragma unroll
    for (int n = 0; n < DSTATE; ++n) {
        Arow[n] = -expf(A_log[(size_t)d * DSTATE + n]);
        h[n] = 0.0f; P[n] = 1.0f;
    }
    __syncthreads();
    float xprev = (c == 0) ? 0.0f : x_ssm[(r0 - 1) * DINNER + d];
#pragma unroll
    for (int n = 0; n < DSTATE; ++n) Bxp[n] = sBp[n] * xprev;

    for (int t = 0; t < CSIZE; ++t) {
        size_t r = r0 + t;
        __syncthreads();
        if (threadIdx.x < 16)      sB[threadIdx.x]      = Bm_rope[r * 16 + threadIdx.x];
        else if (threadIdx.x < 18) sS[threadIdx.x - 16] = scal[r * 2 + threadIdx.x - 16];
        __syncthreads();
        float xv    = x_ssm[r * DINNER + d];
        float delta = softplus_f(sS[0] * wdt + bdt);
        float lam   = sS[1];
#pragma unroll
        for (int n = 0; n < DSTATE; ++n) {
            float alpha = expf(delta * Arow[n]);
            float Bx    = sB[n] * xv;
            float inp   = lam * delta * Bx + (1.0f - lam) * delta * alpha * Bxp[n];
            h[n] = alpha * h[n] + inp;
            P[n] *= alpha;
            Bxp[n] = Bx;
        }
    }
    size_t base = ((((size_t)b * NCHUNK + c) * DINNER) + d) * DSTATE;
#pragma unroll
    for (int n = 0; n < DSTATE; ++n) { Pbuf[base + n] = P[n]; Qbuf[base + n] = h[n]; }
}

__global__ void scan_pass2_kernel(const float* __restrict__ Pbuf,
                                  const float* __restrict__ Qbuf,
                                  float* __restrict__ Hstart) {
    int idx = blockIdx.x * blockDim.x + threadIdx.x;   // BATCH*DINNER*16
    if (idx >= BATCH * DINNER * DSTATE) return;
    int b  = idx / (DINNER * DSTATE);
    int dn = idx % (DINNER * DSTATE);
    float hs = 0.0f;
    for (int c = 0; c < NCHUNK; ++c) {
        size_t i = (((size_t)b * NCHUNK + c) * DINNER * DSTATE) + dn;
        Hstart[i] = hs;
        hs = Pbuf[i] * hs + Qbuf[i];
    }
}

__global__ __launch_bounds__(128) void scan_pass3_kernel(
    const float* __restrict__ x_ssm,
    const float* __restrict__ xz,           // z = cols [DINNER, 2*DINNER)
    const float* __restrict__ Bm_rope,
    const float* __restrict__ Cm_rope,
    const float* __restrict__ scal,
    const float* __restrict__ A_log,
    const float* __restrict__ W_dt,
    const float* __restrict__ b_dt,
    const float* __restrict__ Dp,
    const float* __restrict__ Hstart,
    __bf16* __restrict__ y_bf) {
    int g = blockIdx.x % DGROUPS;
    int c = (blockIdx.x / DGROUPS) % NCHUNK;
    int b = blockIdx.x / (DGROUPS * NCHUNK);
    int d = g * 128 + threadIdx.x;

    __shared__ float sB[16], sC[16], sS[2], sBp[16];
    size_t r0 = (size_t)b * SEQLEN + c * CSIZE;
    if (threadIdx.x < 16)
        sBp[threadIdx.x] = (c == 0) ? 0.0f : Bm_rope[(r0 - 1) * 16 + threadIdx.x];

    float Arow[DSTATE], h[DSTATE], Bxp[DSTATE];
    float wdt = W_dt[d], bdt = b_dt[d], Dd = Dp[d];
    size_t base = ((((size_t)b * NCHUNK + c) * DINNER) + d) * DSTATE;
#pragma unroll
    for (int n = 0; n < DSTATE; ++n) {
        Arow[n] = -expf(A_log[(size_t)d * DSTATE + n]);
        h[n] = Hstart[base + n];
    }
    __syncthreads();
    float xprev = (c == 0) ? 0.0f : x_ssm[(r0 - 1) * DINNER + d];
#pragma unroll
    for (int n = 0; n < DSTATE; ++n) Bxp[n] = sBp[n] * xprev;

    for (int t = 0; t < CSIZE; ++t) {
        size_t r = r0 + t;
        __syncthreads();
        if (threadIdx.x < 16)      sB[threadIdx.x]      = Bm_rope[r * 16 + threadIdx.x];
        else if (threadIdx.x < 32) sC[threadIdx.x - 16] = Cm_rope[r * 16 + threadIdx.x - 16];
        else if (threadIdx.x < 34) sS[threadIdx.x - 32] = scal[r * 2 + threadIdx.x - 32];
        __syncthreads();
        float xv    = x_ssm[r * DINNER + d];
        float delta = softplus_f(sS[0] * wdt + bdt);
        float lam   = sS[1];
        float y = 0.0f;
#pragma unroll
        for (int n = 0; n < DSTATE; ++n) {
            float alpha = expf(delta * Arow[n]);
            float Bx    = sB[n] * xv;
            float inp   = lam * delta * Bx + (1.0f - lam) * delta * alpha * Bxp[n];
            h[n] = alpha * h[n] + inp;
            y   += h[n] * sC[n];
            Bxp[n] = Bx;
        }
        y += xv * Dd;
        float zv = xz[r * N1 + DINNER + d];
        y *= zv * sigmoid_f(zv);
        y_bf[r * DINNER + d] = (__bf16)y;
    }
}

// ---------------------------------------------------------------------------
// Host launcher
// ---------------------------------------------------------------------------
extern "C" void kernel_launch(void* const* d_in, const int* in_sizes, int n_in,
                              void* d_out, int out_size, void* d_ws, size_t ws_size,
                              hipStream_t stream) {
    const float* x       = (const float*)d_in[0];
    const float* W_in    = (const float*)d_in[1];
    const float* conv_w  = (const float*)d_in[2];
    const float* conv_b  = (const float*)d_in[3];
    const float* W_x     = (const float*)d_in[4];
    const float* W_dt    = (const float*)d_in[5];
    const float* b_dt    = (const float*)d_in[6];
    const float* A_log   = (const float*)d_in[7];
    const float* B_bias  = (const float*)d_in[8];
    const float* C_bias  = (const float*)d_in[9];
    const float* W_theta = (const float*)d_in[10];
    const float* W_lam   = (const float*)d_in[11];
    const float* b_lam   = (const float*)d_in[12];
    const float* W_out   = (const float*)d_in[13];
    const float* Dp      = (const float*)d_in[14];
    float* out = (float*)d_out;

    // workspace layout (256B aligned slabs)
    char* ws = (char*)d_ws;
    size_t off = 0;
    auto alloc = [&](size_t bytes) {
        size_t o = off; off += (bytes + 255) & ~(size_t)255; return o;
    };
    __bf16* x_bf    = (__bf16*)(ws + alloc((size_t)NROWS * DMODEL * 2));
    __bf16* W_inT   = (__bf16*)(ws + alloc((size_t)N1 * DMODEL * 2));
    __bf16* W_outT  = (__bf16*)(ws + alloc((size_t)DMODEL * DINNER * 2));
    float*  xz      = (float*) (ws + alloc((size_t)NROWS * N1 * 4));
    float*  x_ssm   = (float*) (ws + alloc((size_t)NROWS * DINNER * 4));
    float*  proj    = (float*) (ws + alloc((size_t)NROWS * PROJW * 4));
    float*  theta_c = (float*) (ws + alloc((size_t)NROWS * 8 * 4));
    float*  Bm_rope = (float*) (ws + alloc((size_t)NROWS * 16 * 4));
    float*  Cm_rope = (float*) (ws + alloc((size_t)NROWS * 16 * 4));
    float*  scal    = (float*) (ws + alloc((size_t)NROWS * 2 * 4));
    __bf16* y_bf    = (__bf16*)(ws + alloc((size_t)NROWS * DINNER * 2));
    const size_t scan_elems = (size_t)BATCH * NCHUNK * DINNER * DSTATE;
    float*  Pbuf    = (float*) (ws + alloc(scan_elems * 4));
    float*  Qbuf    = (float*) (ws + alloc(scan_elems * 4));
    float*  Hstart  = (float*) (ws + alloc(scan_elems * 4));
    (void)ws_size; (void)n_in; (void)in_sizes; (void)out_size;

    // 1) convert / transpose operands for WMMA
    {
        int n = NROWS * DMODEL;
        cvt_bf16_kernel<<<(n + 255) / 256, 256, 0, stream>>>(x, x_bf, n);
    }
    {
        int n = DMODEL * N1;
        transpose_bf16_kernel<<<(n + 255) / 256, 256, 0, stream>>>(W_in, W_inT, DMODEL, N1);
    }
    {
        int n = DINNER * DMODEL;
        transpose_bf16_kernel<<<(n + 255) / 256, 256, 0, stream>>>(W_out, W_outT, DINNER, DMODEL);
    }

    // 2) xz = x @ W_in  (WMMA, 16x64 strip per wave)
    {
        int waves  = (NROWS / 16) * (N1 / 64);
        int blocks = (waves * 32 + 255) / 256;
        gemm_bf16_wmma_kernel<<<blocks, 256, 0, stream>>>(x_bf, W_inT, xz,
                                                          NROWS, N1, DMODEL);
    }

    // 3) causal depthwise conv + SiLU
    {
        int n = NROWS * DINNER;
        conv_silu_kernel<<<(n + 255) / 256, 256, 0, stream>>>(xz, conv_w, conv_b, x_ssm);
    }

    // 4) fused small projection (W_x | W_theta | W_lam)
    proj_kernel<<<NROWS, 64, 0, stream>>>(x_ssm, W_x, W_theta, W_lam, proj);

    // 5) theta cumsum along L
    cumsum_theta_kernel<<<BATCH, 8, 0, stream>>>(proj, theta_c);

    // 6) RoPE + scalars
    rope_kernel<<<(NROWS + 127) / 128, 128, 0, stream>>>(
        proj, theta_c, B_bias, C_bias, b_lam, Bm_rope, Cm_rope, scal);

    // 7) chunked parallel recurrence
    scan_pass1_kernel<<<BATCH * NCHUNK * DGROUPS, 128, 0, stream>>>(
        x_ssm, Bm_rope, scal, A_log, W_dt, b_dt, Pbuf, Qbuf);
    {
        int n = BATCH * DINNER * DSTATE;
        scan_pass2_kernel<<<(n + 255) / 256, 256, 0, stream>>>(Pbuf, Qbuf, Hstart);
    }
    scan_pass3_kernel<<<BATCH * NCHUNK * DGROUPS, 128, 0, stream>>>(
        x_ssm, xz, Bm_rope, Cm_rope, scal, A_log, W_dt, b_dt, Dp, Hstart, y_bf);

    // 8) out = y @ W_out  (WMMA)
    {
        int waves  = (NROWS / 16) * (DMODEL / 64);
        int blocks = (waves * 32 + 255) / 256;
        gemm_bf16_wmma_kernel<<<blocks, 256, 0, stream>>>(y_bf, W_outT, out,
                                                          NROWS, DMODEL, DINNER);
    }
}
